// ThreeBodyAggregation_38732015075509
// MI455X (gfx1250) — compile-verified, matrix operand order checked by reading, not run on hardware
//
#include <hip/hip_runtime.h>
#include <hip/hip_bf16.h>

typedef __attribute__((ext_vector_type(16))) _Float16 v16h;
typedef __attribute__((ext_vector_type(8)))  float    v8f;

#define B_    4
#define N_    160
#define R_    64
#define HID_  128
#define EMB_  128
#define NT_   (N_ / 16)        // 10 tiles per dimension
#define NTILES_ (NT_ * NT_)    // 100 tiles of 16x16

// One block per (b,i). 256 threads = 8 waves (wave32).
//  - stage r_hat[b,i,:,0:3] into LDS via async global->LDS copies (ASYNCcnt),
//    overlapped with the radial-gate compute which streams rbf from HBM
//  - WMMA tile loop: cos-theta Gram tiles via v_wmma_f32_16x16x32_f16
//    (A: M=lane(0..15), K=0..2 = r_hat components, rest zero;
//     B: N=lane(0..15), K=0..2 = r_hat components, rest zero;
//     C/D layout: lane%16 = N, vgpr v = M row, lanes>=16 -> M+8)
//  - clip + Legendre P0..P3 + w_j*w_k weighting on the f32 accumulator
//  - shuffle-reduce per wave, ds_add_f32 into 4 LDS accumulators
//  - fused MLP (4->128 silu, 128->128) + LayerNorm tail
__launch_bounds__(256)
__global__ void three_body_agg_kernel(
    const float* __restrict__ rbf,      // (B,N,N,R)
    const float* __restrict__ r_hat,    // (B,N,N,3)
    const float* __restrict__ mask,     // (B,N,N)
    const float* __restrict__ w_rad,    // (1,R)
    const float* __restrict__ b_rad,    // (1,)
    const float* __restrict__ w1,       // (HID,4)
    const float* __restrict__ b1,       // (HID,)
    const float* __restrict__ w2,       // (EMB,HID)
    const float* __restrict__ b2,       // (EMB,)
    const float* __restrict__ gamma,    // (EMB,)
    const float* __restrict__ beta,     // (EMB,)
    float* __restrict__ out)            // (B,N,EMB)
{
    __shared__ float lds_rh[N_ * 3];
    __shared__ float lds_w[N_];
    __shared__ float lds_acc[4];
    __shared__ float lds_h1[HID_];
    __shared__ float lds_rs[8], lds_rq[8];
    __shared__ float lds_mu, lds_rstd;

    const int tid  = threadIdx.x;
    const int lane = tid & 31;
    const int wid  = tid >> 5;
    const int blk  = blockIdx.x;              // flat (b*N + i)

    // ---- async stage of r_hat row block into LDS (480 contiguous floats) ----
    // GLOBAL_LOAD_ASYNC_TO_LDS_B32 (GV mode): copies one dword from vaddr to LDS at vdst; tracked
    // by ASYNCcnt, so the radial-gate rbf streaming below overlaps the copy.
    const float* rh_src = r_hat + (size_t)blk * N_ * 3;
    for (int idx = tid; idx < N_ * 3; idx += 256) {
        unsigned int  lds_addr = (unsigned int)(size_t)&lds_rh[idx];
        unsigned long gaddr    = (unsigned long)(size_t)(rh_src + idx);
        asm volatile("global_load_async_to_lds_b32 %0, %1, off"
                     :: "v"(lds_addr), "v"(gaddr) : "memory");
    }
    if (tid < 4) lds_acc[tid] = 0.0f;

    // ---- radial gate: w[j] = sigmoid(rbf[b,i,j,:]·w_rad + b_rad) * mask ----
    if (tid < N_) {
        const float4* rb = (const float4*)(rbf + ((size_t)blk * N_ + tid) * R_);
        const float4* wr = (const float4*)w_rad;
        float acc = 0.0f;
        #pragma unroll
        for (int q = 0; q < R_ / 4; ++q) {
            __builtin_prefetch(rb + q + 16, 0, 1);   // global_prefetch_b8
            float4 v = rb[q], u = wr[q];
            acc += v.x * u.x + v.y * u.y + v.z * u.z + v.w * u.w;
        }
        acc += b_rad[0];
        float sg = 1.0f / (1.0f + __expf(-acc));
        lds_w[tid] = sg * mask[(size_t)blk * N_ + tid];
    }

    // drain the async LDS copies, then make them visible block-wide
    asm volatile("s_wait_asynccnt 0" ::: "memory");
    __syncthreads();

    // ---- WMMA tile loop over (j,k) 16x16 tiles ----
    float s0 = 0.f, s1 = 0.f, s2 = 0.f, s3 = 0.f;
    for (int t = wid; t < NTILES_; t += 8) {
        const int j0 = (t / NT_) * 16;
        const int k0 = (t % NT_) * 16;

        v16h A  = {};
        v16h Bm = {};
        if (lane < 16) {
            const int j = j0 + lane;              // A row M = lane, K=0..2
            A[0] = (_Float16)lds_rh[3 * j + 0];
            A[1] = (_Float16)lds_rh[3 * j + 1];
            A[2] = (_Float16)lds_rh[3 * j + 2];
            const int k = k0 + lane;              // B col N = lane, K=0..2
            Bm[0] = (_Float16)lds_rh[3 * k + 0];
            Bm[1] = (_Float16)lds_rh[3 * k + 1];
            Bm[2] = (_Float16)lds_rh[3 * k + 2];
        }
        const float wcol = lds_w[k0 + (lane & 15)];

        v8f c = {};
        c = __builtin_amdgcn_wmma_f32_16x16x32_f16(
                /*neg_a=*/false, A, /*neg_b=*/false, Bm,
                /*c_mod=*/(short)0, c, /*reuse_a=*/false, /*reuse_b=*/false);

        const int mbase = j0 + ((lane >= 16) ? 8 : 0);
        #pragma unroll
        for (int v = 0; v < 8; ++v) {
            float x = c[v];                                   // cos theta
            x = fminf(fmaxf(x, -1.0f + 1e-6f), 1.0f - 1e-6f); // clip
            const float ww = lds_w[mbase + v] * wcol;         // w_j * w_k
            const float x2 = x * x;
            s0 += ww;                           // P0 = 1
            s1 += ww * x;                       // P1 = x
            s2 += ww * (1.5f * x2 - 0.5f);      // P2
            s3 += ww * ((2.5f * x2 - 1.5f) * x);// P3
        }
    }

    // wave32 shuffle reduction, then LDS float atomics (ds_add_f32)
    #pragma unroll
    for (int m = 16; m; m >>= 1) {
        s0 += __shfl_xor(s0, m);
        s1 += __shfl_xor(s1, m);
        s2 += __shfl_xor(s2, m);
        s3 += __shfl_xor(s3, m);
    }
    if (lane == 0) {
        atomicAdd(&lds_acc[0], s0);
        atomicAdd(&lds_acc[1], s1);
        atomicAdd(&lds_acc[2], s2);
        atomicAdd(&lds_acc[3], s3);
    }
    __syncthreads();

    // ---- fused MLP: h1 = silu(agg @ w1^T + b1) ----
    if (tid < HID_) {
        const float a0 = lds_acc[0], a1 = lds_acc[1];
        const float a2 = lds_acc[2], a3 = lds_acc[3];
        float p = b1[tid] + a0 * w1[tid * 4 + 0] + a1 * w1[tid * 4 + 1]
                          + a2 * w1[tid * 4 + 2] + a3 * w1[tid * 4 + 3];
        lds_h1[tid] = p / (1.0f + __expf(-p));   // silu
    }
    __syncthreads();

    // ---- h2 = h1 @ w2^T + b2 ----
    float h2v = 0.0f;
    if (tid < EMB_) {
        const float* w2r = w2 + (size_t)tid * HID_;
        float acc = b2[tid];
        #pragma unroll 4
        for (int h = 0; h < HID_; ++h) acc += lds_h1[h] * w2r[h];
        h2v = acc;
    }

    // ---- LayerNorm over EMB: two-level reduction ----
    float rs = h2v, rq = h2v * h2v;     // inactive lanes contribute zeros
    #pragma unroll
    for (int m = 16; m; m >>= 1) {
        rs += __shfl_xor(rs, m);
        rq += __shfl_xor(rq, m);
    }
    if (lane == 0) { lds_rs[wid] = rs; lds_rq[wid] = rq; }
    __syncthreads();
    if (tid == 0) {
        float S = 0.f, Q = 0.f;
        #pragma unroll
        for (int w = 0; w < 8; ++w) { S += lds_rs[w]; Q += lds_rq[w]; }
        const float mu  = S / (float)EMB_;
        const float var = Q / (float)EMB_ - mu * mu;
        lds_mu   = mu;
        lds_rstd = rsqrtf(var + 1e-5f);
    }
    __syncthreads();
    if (tid < EMB_) {
        out[(size_t)blk * EMB_ + tid] =
            (h2v - lds_mu) * lds_rstd * gamma[tid] + beta[tid];
    }
}

extern "C" void kernel_launch(void* const* d_in, const int* in_sizes, int n_in,
                              void* d_out, int out_size, void* d_ws, size_t ws_size,
                              hipStream_t stream) {
    (void)in_sizes; (void)n_in; (void)d_ws; (void)ws_size; (void)out_size;
    const float* rbf    = (const float*)d_in[0];
    const float* r_hat  = (const float*)d_in[1];
    const float* mask   = (const float*)d_in[2];
    const float* w_rad  = (const float*)d_in[3];
    const float* b_rad  = (const float*)d_in[4];
    const float* w1     = (const float*)d_in[5];
    const float* b1     = (const float*)d_in[6];
    const float* w2     = (const float*)d_in[7];
    const float* b2     = (const float*)d_in[8];
    const float* gamma  = (const float*)d_in[9];
    const float* beta   = (const float*)d_in[10];
    float* out = (float*)d_out;

    dim3 grid(B_ * N_);   // 640 blocks: one per (b, i)
    dim3 block(256);      // 8 waves of 32
    three_body_agg_kernel<<<grid, block, 0, stream>>>(
        rbf, r_hat, mask, w_rad, b_rad, w1, b1, w2, b2, gamma, beta, out);
}
